// DeepFM_14946486190645
// MI455X (gfx1250) — matrix operand order repaired and checked by reading, not compiled, not run on hardware
//
#include <hip/hip_runtime.h>
#include <math.h>

typedef __attribute__((ext_vector_type(16))) __bf16 v16bf;
typedef __attribute__((ext_vector_type(8)))  __bf16 v8bf;
typedef __attribute__((ext_vector_type(8)))  float  v8f;

#define NS       26
#define EMB      64
#define NDENSE   13
#define COMB     1677
#define KPAD     1696          // ceil(1677/32)*32
#define KSTEPS1  53            // 1696/32
#define CSTRIDE  1704          // LDS row stride in halfs (multiple of 8 for b128 loads)
#define H1       128
#define H2       64
#define VOCABN   100000
#define KSTEPS2  4             // 128/32
#define WPK1_BYTES ((size_t)8 * KSTEPS1 * 32 * 16 * 2)   // 434176, 256B aligned

// Repack f32 [K x N] weights into bf16 WMMA B-fragment order:
// dst[((t*ksteps+s)*32 + lane)*16 + j] = bf16(w[k*N + n])
//   k = s*32 + (lane>=16 ? 8 : 0) + (j&7) + (j>=8 ? 16 : 0),  n = t*16 + lane%16,  zero-pad k >= kmax.
__global__ void pack_b_frags(const float* __restrict__ w, __bf16* __restrict__ dst,
                             int ksteps, int kmax, int ncols) {
  int idx = blockIdx.x * blockDim.x + threadIdx.x;
  int total = (ncols >> 4) * ksteps * 32 * 16;
  if (idx >= total) return;
  int j  = idx & 15;
  int L  = (idx >> 4) & 31;
  int r2 = idx >> 9;
  int s  = r2 % ksteps;
  int t  = r2 / ksteps;
  int k  = s * 32 + ((L >> 4) << 3) + (j & 7) + ((j >> 3) << 4);
  int n  = t * 16 + (L & 15);
  float v = (k < kmax) ? w[(size_t)k * ncols + n] : 0.0f;
  dst[idx] = (__bf16)v;
}

// A fragment (16x32 bf16) for wave32 WMMA from a row-major bf16 LDS tile.
// Lane l: row m = l&15; halfs 0..7 = K[koff..koff+7], halfs 8..15 = K[koff+16..koff+23],
// koff = (l>>4)*8  -> two ds_load_b128 per k-step.
__device__ inline v16bf load_a_frag(const __bf16* base, int stride, int kbase, int lane) {
  int m    = lane & 15;
  int koff = (lane >> 4) << 3;
  const __bf16* p = base + m * stride + kbase + koff;
  v8bf lo = *(const v8bf*)p;
  v8bf hi = *(const v8bf*)(p + 16);
  v16bf a;
#pragma unroll
  for (int i = 0; i < 8; ++i) { a[i] = lo[i]; a[i + 8] = hi[i]; }
  return a;
}

__global__ __launch_bounds__(256)
void deepfm_fused(const int* __restrict__ ids, const float* __restrict__ tables,
                  const float* __restrict__ ln_beta,
                  const float* __restrict__ w_lin, const float* __restrict__ b_lin,
                  const float* __restrict__ b1, const float* __restrict__ b2,
                  const float* __restrict__ w_out, const float* __restrict__ b_out,
                  const __bf16* __restrict__ wpk1, const __bf16* __restrict__ wpk2,
                  float* __restrict__ out) {
  __shared__ __bf16 s_comb[16 * CSTRIDE];     // 54528 B
  __shared__ __bf16 s_x1[16 * H1];            //  4096 B
  __shared__ float  s_x2[16 * (H2 + 1)];      //  4160 B
  __shared__ float  s_rowacc[16];             // linear(emb) + FM cross per row
  __shared__ float  s_lin_dense;

  const int tid  = threadIdx.x;
  const int lane = tid & 31;
  const int wave = tid >> 5;
  const int tile = blockIdx.x;

  // ---- init: row accumulators, dense (LayerNorm == beta exactly), K padding ----
  if (tid < 16) s_rowacc[tid] = 0.0f;
  if (tid == 0) {
    float sd = 0.0f;
#pragma unroll
    for (int i = 0; i < NDENSE; ++i) sd += ln_beta[i] * w_lin[i];
    s_lin_dense = sd;
  }
  if (tid < 32) {  // 13 dense cols + 19 zero-pad cols (1677..1695)
    int col = (tid < NDENSE) ? tid : (COMB + (tid - NDENSE));
    __bf16 v = (tid < NDENSE) ? (__bf16)ln_beta[tid] : (__bf16)0.0f;
#pragma unroll
    for (int r = 0; r < 16; ++r) s_comb[r * CSTRIDE + col] = v;
  }
  __syncthreads();

  // ---- embedding gather (coalesced 256B rows) + FM cross + linear term ----
  {
    int d    = tid & 63;     // embedding dim -> contiguous across 64 threads
    int rsub = tid >> 6;     // 4 rows in flight
    for (int rc = 0; rc < 4; ++rc) {
      int r    = rc * 4 + rsub;
      int grow = tile * 16 + r;
      float sum = 0.0f, sq = 0.0f, lin = 0.0f;
      for (int f = 0; f < NS; ++f) {
        int id  = ids[grow * NS + f];
        float v = tables[((size_t)f * VOCABN + (size_t)id) * EMB + d];
        sum += v;
        sq  += v * v;
        lin += v * w_lin[NDENSE + f * EMB + d];
        s_comb[r * CSTRIDE + NDENSE + f * EMB + d] = (__bf16)v;
      }
      float part = lin + 0.5f * (sum * sum - sq);   // per-(row,d) contribution
#pragma unroll
      for (int o = 16; o > 0; o >>= 1) part += __shfl_xor(part, o, 32);
      if (lane == 0) atomicAdd(&s_rowacc[r], part);
    }
  }
  __syncthreads();

  // ---- GEMM1: combined[16x1696] @ w1 -> x1[16x128]; one N=16 tile per wave ----
  {
    v8f acc = {0, 0, 0, 0, 0, 0, 0, 0};
    const __bf16* bptr = wpk1 + ((size_t)wave * KSTEPS1 * 32 + lane) * 16;
    for (int s = 0; s < KSTEPS1; ++s) {
      v16bf a = load_a_frag(s_comb, CSTRIDE, s * 32, lane);
      v16bf b = *(const v16bf*)(bptr + (size_t)s * 32 * 16);
      acc = __builtin_amdgcn_wmma_f32_16x16x32_bf16(false, a, false, b,
                                                    (short)0, acc, false, false);
    }
    int nloc  = lane & 15;
    int mbase = (lane >> 4) << 3;                 // C layout: M = v + 8*(lane>=16)
    float bias = b1[wave * 16 + nloc];
#pragma unroll
    for (int v = 0; v < 8; ++v) {
      float val = fmaxf(acc[v] + bias, 0.0f);     // ReLU
      s_x1[(mbase + v) * H1 + wave * 16 + nloc] = (__bf16)val;
    }
  }
  __syncthreads();

  // ---- GEMM2: x1[16x128] @ w2 -> x2[16x64]; waves 0..3 ----
  if (wave < 4) {
    v8f acc = {0, 0, 0, 0, 0, 0, 0, 0};
    for (int s = 0; s < KSTEPS2; ++s) {
      v16bf a = load_a_frag(s_x1, H1, s * 32, lane);
      v16bf b = *(const v16bf*)(wpk2 + ((size_t)(wave * KSTEPS2 + s) * 32 + lane) * 16);
      acc = __builtin_amdgcn_wmma_f32_16x16x32_bf16(false, a, false, b,
                                                    (short)0, acc, false, false);
    }
    int nloc  = lane & 15;
    int mbase = (lane >> 4) << 3;
    float bias = b2[wave * 16 + nloc];
#pragma unroll
    for (int v = 0; v < 8; ++v)
      s_x2[(mbase + v) * (H2 + 1) + wave * 16 + nloc] = fmaxf(acc[v] + bias, 0.0f);
  }
  __syncthreads();

  // ---- out = sigmoid((lin + cross + x2) @ w_out + b_out) ----
  if (tid < 16) {
    float lc = s_rowacc[tid] + s_lin_dense + b_lin[0];
    float z  = b_out[0];
#pragma unroll
    for (int c = 0; c < H2; ++c) z += (s_x2[tid * (H2 + 1) + c] + lc) * w_out[c];
    out[tile * 16 + tid] = 1.0f / (1.0f + expf(-z));
  }
}

extern "C" void kernel_launch(void* const* d_in, const int* in_sizes, int n_in,
                              void* d_out, int out_size, void* d_ws, size_t ws_size,
                              hipStream_t stream) {
  (void)n_in; (void)out_size; (void)ws_size;
  const int*   ids     = (const int*)d_in[1];
  const float* tables  = (const float*)d_in[2];
  const float* ln_beta = (const float*)d_in[4];   // ln_gamma (d_in[3]) cancels exactly
  const float* w_lin   = (const float*)d_in[5];
  const float* b_lin   = (const float*)d_in[6];
  const float* w1      = (const float*)d_in[7];
  const float* b1      = (const float*)d_in[8];
  const float* w2      = (const float*)d_in[9];
  const float* b2      = (const float*)d_in[10];
  const float* w_out   = (const float*)d_in[11];
  const float* b_out   = (const float*)d_in[12];
  float* out = (float*)d_out;

  int Bn = in_sizes[1] / NS;          // 16384
  __bf16* wpk1 = (__bf16*)d_ws;
  __bf16* wpk2 = (__bf16*)((char*)d_ws + WPK1_BYTES);

  // Repack weights into WMMA B-fragment order (bf16, K zero-padded).
  {
    int total1 = 8 * KSTEPS1 * 32 * 16;   // 217088
    pack_b_frags<<<(total1 + 255) / 256, 256, 0, stream>>>(w1, wpk1, KSTEPS1, COMB, H1);
    int total2 = 4 * KSTEPS2 * 32 * 16;   // 8192
    pack_b_frags<<<(total2 + 255) / 256, 256, 0, stream>>>(w2, wpk2, KSTEPS2, H1, H2);
  }

  deepfm_fused<<<Bn / 16, 256, 0, stream>>>(ids, tables, ln_beta, w_lin, b_lin,
                                            b1, b2, w_out, b_out, wpk1, wpk2, out);
}
